// LLamaAttention_27410481283331
// MI455X (gfx1250) — matrix operand order repaired and checked by reading, not compiled, not run on hardware
//
#include <hip/hip_runtime.h>
#include <hip/hip_bf16.h>

#define B_    2
#define S_    2048
#define H_    2048
#define NH_   16
#define NKV_  4
#define HD_   128
#define GROUPS_ (NH_ / NKV_)
#define NEG_  (-1000000000.0f)
#define ROTBASE_ 10000.0f

typedef __attribute__((ext_vector_type(16))) __bf16 v16bf;
typedef __attribute__((ext_vector_type(8)))  float  v8f;

union BF16Frag { v16bf v; unsigned short u[16]; };

__device__ __forceinline__ unsigned short f2b(float f) {
    unsigned u = __builtin_bit_cast(unsigned, f);
    unsigned r = u + 0x7FFFu + ((u >> 16) & 1u);
    return (unsigned short)(r >> 16);
}
__device__ __forceinline__ float b2f(unsigned short h) {
    unsigned u = ((unsigned)h) << 16;
    return __builtin_bit_cast(float, u);
}

// Two contiguous 16B LDS reads -> one 16-element bf16 fragment.
__device__ __forceinline__ void frag_ld(BF16Frag& f, const unsigned short* p0,
                                        const unsigned short* p1) {
    *(uint4*)&f.u[0] = *(const uint4*)p0;
    *(uint4*)&f.u[8] = *(const uint4*)p1;
}

// gfx1250 async global->LDS copy of 16 bytes (tracked with ASYNCcnt).
__device__ __forceinline__ void async_cp16(void* lds_ptr, const void* gptr) {
    unsigned lds_off = (unsigned)(unsigned long long)lds_ptr;
    asm volatile("global_load_async_to_lds_b128 %0, %1, off"
                 :: "v"(lds_off), "v"(gptr) : "memory");
}
__device__ __forceinline__ void async_wait0() {
    asm volatile("s_wait_asynccnt 0" ::: "memory");
}

// ---------------------------------------------------------------------------
// f32 -> bf16 convert (row-major copy)
// ---------------------------------------------------------------------------
__global__ void cvt_f32_bf16(const float* __restrict__ src,
                             unsigned short* __restrict__ dst, size_t n) {
    size_t i = (size_t)blockIdx.x * blockDim.x + threadIdx.x;
    if (i < n) dst[i] = f2b(src[i]);
}

// ---------------------------------------------------------------------------
// f32 [K][N] -> bf16 [N][K] tiled transpose+convert (one-time weight prep).
// Block 256 threads = 32x8, 32x32 tile via LDS (pitch 33: conflict-free).
// ---------------------------------------------------------------------------
__global__ __launch_bounds__(256)
void cvt_transpose(const float* __restrict__ src, unsigned short* __restrict__ dst,
                   int K, int N) {
    __shared__ float tile[32][33];
    int n0 = blockIdx.x * 32, k0 = blockIdx.y * 32;
    int tx = threadIdx.x & 31, ty = threadIdx.x >> 5;
    for (int i = 0; i < 4; i++)
        tile[ty + i * 8][tx] = src[(size_t)(k0 + ty + i * 8) * N + n0 + tx];
    __syncthreads();
    for (int i = 0; i < 4; i++)
        dst[(size_t)(n0 + ty + i * 8) * K + k0 + tx] = f2b(tile[tx][ty + i * 8]);
}

// ---------------------------------------------------------------------------
// bf16 GEMM: C[M,N] = A[M,K] @ W[K,N], weights given PRE-TRANSPOSED Wt[N][K].
// 128x128 block tile, 256 threads = 8 waves, wave = 32x64 subtile
// (2x4 wmma_f32_16x16x32_bf16).  Both tiles staged with async global->LDS
// b128 copies, double-buffered on ASYNCcnt so fetch overlaps WMMA.
// TOUT=true stores C transposed ([N][M]) with packed 16B stores.
// ---------------------------------------------------------------------------
template <bool F32OUT, bool TOUT>
__global__ __launch_bounds__(256)
void gemm_bf16(const unsigned short* __restrict__ A,
               const unsigned short* __restrict__ Wt,
               void* __restrict__ C, int M, int N, int K) {
    __shared__ unsigned short As[2][128][40];  // [buf][m][k], pitch 80B
    __shared__ unsigned short Bt[2][128][40];  // [buf][n][k], pitch 80B

    const int tid  = threadIdx.x;
    const int wave = tid >> 5, lane = tid & 31;
    const int wr = wave >> 1, wc = wave & 1;
    const int m0 = blockIdx.y * 128, n0 = blockIdx.x * 128;
    const int lm  = lane & 15;
    const int kbA = (lane < 16) ? 0 : 8;    // 16-bit A-matrix layout
    const int kbB = (lane < 16) ? 0 : 8;    // 16-bit B-matrix layout
    const int rb  = (lane < 16) ? 0 : 8;    // 32-bit C/D layout

    v8f acc[2][4] = {};

    auto stage = [&](int p, int k0) {
        for (int c = tid; c < 512; c += 256) {
            int r = c >> 2, cc = (c & 3) << 3;
            async_cp16(&As[p][r][cc], A  + (size_t)(m0 + r) * K + k0 + cc);
        }
        for (int c = tid; c < 512; c += 256) {
            int r = c >> 2, cc = (c & 3) << 3;
            async_cp16(&Bt[p][r][cc], Wt + (size_t)(n0 + r) * K + k0 + cc);
        }
    };

    stage(0, 0);
    async_wait0();
    __syncthreads();

    int p = 0;
    for (int k0 = 0; k0 < K; k0 += 32) {
        if (k0 + 32 < K) stage(p ^ 1, k0 + 32);  // prefetch next tile (async)

        BF16Frag af[2], bfr[4];
        for (int i = 0; i < 2; i++) {
            int ar = wr * 32 + i * 16 + lm;
            frag_ld(af[i], &As[p][ar][kbA], &As[p][ar][kbA + 16]);
        }
        for (int j = 0; j < 4; j++) {
            int bc = wc * 64 + j * 16 + lm;
            frag_ld(bfr[j], &Bt[p][bc][kbB], &Bt[p][bc][kbB + 16]);
        }
        for (int i = 0; i < 2; i++)
            for (int j = 0; j < 4; j++)
                acc[i][j] = __builtin_amdgcn_wmma_f32_16x16x32_bf16(
                    false, af[i].v, false, bfr[j].v, (short)0, acc[i][j],
                    false, false);

        async_wait0();
        __syncthreads();
        p ^= 1;
    }

    for (int i = 0; i < 2; i++)
        for (int j = 0; j < 4; j++) {
            int col = n0 + wc * 64 + j * 16 + lm;
            int row0 = m0 + wr * 32 + i * 16 + rb;
            if (TOUT) {
                // C[col][row]: rows e=0..7 contiguous -> one 16B packed store
                uint4 pk;
                unsigned* w = (unsigned*)&pk;
                for (int e = 0; e < 4; e++)
                    w[e] = (unsigned)f2b(acc[i][j][2 * e]) |
                           ((unsigned)f2b(acc[i][j][2 * e + 1]) << 16);
                *(uint4*)&((unsigned short*)C)[(size_t)col * M + row0] = pk;
            } else {
                for (int e = 0; e < 8; e++) {
                    int row = row0 + e;
                    if (F32OUT)
                        ((float*)C)[(size_t)row * N + col] = acc[i][j][e];
                    else
                        ((unsigned short*)C)[(size_t)row * N + col] = f2b(acc[i][j][e]);
                }
            }
        }
}

// ---------------------------------------------------------------------------
// In-place RoPE on bf16 tensor laid out [B, S, nheads*HD].
// ---------------------------------------------------------------------------
__global__ void rope_kernel(unsigned short* __restrict__ T,
                            const int* __restrict__ pos_ids, int nheads) {
    size_t idx   = (size_t)blockIdx.x * blockDim.x + threadIdx.x;
    size_t total = (size_t)B_ * S_ * nheads * (HD_ / 2);
    if (idx >= total) return;
    int    d  = (int)(idx % (HD_ / 2));
    size_t t1 = idx / (HD_ / 2);
    int    hh = (int)(t1 % nheads);
    size_t t2 = t1 / nheads;
    int    s  = (int)(t2 % S_);
    int    bb = (int)(t2 / S_);

    int   pos = pos_ids[(size_t)bb * S_ + s];
    float inv_freq = __powf(ROTBASE_, -((float)(2 * d) / (float)HD_));
    float ang = (float)pos * inv_freq;
    float sn = __sinf(ang), cs = __cosf(ang);

    size_t base = (((size_t)bb * S_ + s) * nheads + hh) * HD_;
    float a  = b2f(T[base + d]);
    float bq = b2f(T[base + d + HD_ / 2]);
    T[base + d]           = f2b(a * cs - bq * sn);
    T[base + d + HD_ / 2] = f2b(bq * cs + a * sn);
}

// ---------------------------------------------------------------------------
// Flash-style causal GQA attention.
// Grid: (S/64, NH, B). Block: 128 threads = 4 waves; wave owns 16 Q rows.
// K kept [key][d] (QK^T B-frags contiguous); V arrives PRE-TRANSPOSED in
// global ([NKV*HD][B*S]) so the [d][key] tile stages as contiguous async
// b128 copies and P@V B-frags are contiguous b128 reads.
// ---------------------------------------------------------------------------
__global__ __launch_bounds__(128)
void attn_kernel(const unsigned short* __restrict__ Qb,
                 const unsigned short* __restrict__ Kb,
                 const unsigned short* __restrict__ Vtg,
                 unsigned short* __restrict__ Ob) {
    __shared__ unsigned short Ks[64][136];   // [key][d], pitch 272B (Q staging too)
    __shared__ unsigned short Vt[128][72];   // [d][key], pitch 144B
    __shared__ float          Sc[64][68];    // scores, pitch 272B
    __shared__ unsigned short Pb[64][72];    // softmax numerators, pitch 144B
    __shared__ float rowM[64], rowL[64], rowF[64];

    const int tid  = threadIdx.x;
    const int wave = tid >> 5, lane = tid & 31;
    const int b    = blockIdx.z, h = blockIdx.y;
    const int q0   = blockIdx.x * 64;
    const int kvh  = h / GROUPS_;
    const int lm   = lane & 15;
    const int kbA  = (lane < 16) ? 0 : 8;
    const int kbB  = (lane < 16) ? 0 : 8;
    const int rb   = (lane < 16) ? 0 : 8;
    const float scale = 0.08838834764831845f;  // 1/sqrt(128)

    // ---- Stage Q tile (64 x 128) via async copies, build fragments.
    for (int c = tid; c < 1024; c += 128) {
        int r = c >> 4, cc = (c & 15) << 3;
        async_cp16(&Ks[r][cc],
                   Qb + ((size_t)(b * S_ + q0 + r) * (NH_ * HD_)) + h * HD_ + cc);
    }
    async_wait0();
    __syncthreads();
    BF16Frag qf[4];
    {
        int row = wave * 16 + lm;
        for (int w4 = 0; w4 < 4; w4++)
            frag_ld(qf[w4], &Ks[row][w4 * 32 + kbA], &Ks[row][w4 * 32 + kbA + 16]);
    }
    if (tid < 64) { rowM[tid] = -3.0e38f; rowL[tid] = 0.f; }
    __syncthreads();

    v8f o[8] = {};  // wave's 16 rows x 128 cols (8 n-tiles)

    for (int kb0 = 0; kb0 <= q0; kb0 += 64) {
        // ---- Stage K [key][d] and V^T [d][key] tiles, both pure async b128.
        for (int c = tid; c < 1024; c += 128) {
            int r = c >> 4, cc = (c & 15) << 3;
            async_cp16(&Ks[r][cc],
                       Kb + ((size_t)(b * S_ + kb0 + r) * (NKV_ * HD_)) + kvh * HD_ + cc);
        }
        for (int c = tid; c < 1024; c += 128) {
            int d = c >> 3, cc = (c & 7) << 3;
            async_cp16(&Vt[d][cc],
                       Vtg + (size_t)(kvh * HD_ + d) * (B_ * S_) + b * S_ + kb0 + cc);
        }
        async_wait0();
        __syncthreads();

        // ---- Scores: wave's 16 q rows x 64 keys (4 n-tiles x 4 k-steps)
        v8f sacc[4] = {};
        for (int j = 0; j < 4; j++) {
            int n = j * 16 + lm;  // key index
            for (int w4 = 0; w4 < 4; w4++) {
                BF16Frag bk;
                frag_ld(bk, &Ks[n][w4 * 32 + kbB], &Ks[n][w4 * 32 + kbB + 16]);
                sacc[j] = __builtin_amdgcn_wmma_f32_16x16x32_bf16(
                    false, qf[w4].v, false, bk.v, (short)0, sacc[j], false, false);
            }
        }
        // ---- Scale + causal mask, spill to LDS
        for (int j = 0; j < 4; j++) {
            int col  = j * 16 + lm;
            int kpos = kb0 + col;
            for (int e = 0; e < 8; e++) {
                int r = wave * 16 + rb + e;
                float s = sacc[j][e] * scale;
                if (kpos > q0 + r) s += NEG_;
                Sc[r][col] = s;
            }
        }
        __syncthreads();

        // ---- Online softmax (one thread per q row, float4 row scans)
        if (tid < 64) {
            float mprev = rowM[tid];
            float mx = mprev;
            const float4* rowp = (const float4*)&Sc[tid][0];
            for (int c4 = 0; c4 < 16; c4++) {
                float4 v4 = rowp[c4];
                mx = fmaxf(mx, fmaxf(fmaxf(v4.x, v4.y), fmaxf(v4.z, v4.w)));
            }
            float fac = __expf(mprev - mx);
            float sum = 0.f;
            for (int c4 = 0; c4 < 16; c4++) {
                float4 v4 = rowp[c4];
                float p0 = __expf(v4.x - mx), p1 = __expf(v4.y - mx);
                float p2 = __expf(v4.z - mx), p3 = __expf(v4.w - mx);
                sum += (p0 + p1) + (p2 + p3);
                uint2 pk;
                pk.x = (unsigned)f2b(p0) | ((unsigned)f2b(p1) << 16);
                pk.y = (unsigned)f2b(p2) | ((unsigned)f2b(p3) << 16);
                *(uint2*)&Pb[tid][c4 * 4] = pk;
            }
            rowM[tid] = mx;
            rowL[tid] = rowL[tid] * fac + sum;
            rowF[tid] = fac;
        }
        __syncthreads();

        // ---- Rescale running accumulators, O += P (16x64) @ V (64x128)
        for (int e = 0; e < 8; e++) {
            float fac = rowF[wave * 16 + rb + e];
            for (int j = 0; j < 8; j++) o[j][e] *= fac;
        }
        {
            int row = wave * 16 + lm;
            for (int w2 = 0; w2 < 2; w2++) {
                BF16Frag pa;
                frag_ld(pa, &Pb[row][w2 * 32 + kbA], &Pb[row][w2 * 32 + kbA + 16]);
                for (int j = 0; j < 8; j++) {
                    BF16Frag bv;
                    int nd = j * 16 + lm;  // output head-dim column
                    frag_ld(bv, &Vt[nd][w2 * 32 + kbB], &Vt[nd][w2 * 32 + kbB + 16]);
                    o[j] = __builtin_amdgcn_wmma_f32_16x16x32_bf16(
                        false, pa.v, false, bv.v, (short)0, o[j], false, false);
                }
            }
        }
        __syncthreads();
    }

    // ---- Normalize by row sums and emit bf16 [B, S, NH*HD]
    for (int e = 0; e < 8; e++) {
        int r = wave * 16 + rb + e;
        float inv = 1.0f / rowL[r];
        size_t srow = (size_t)(b * S_ + q0 + r);
        for (int j = 0; j < 8; j++) {
            int col = j * 16 + lm;
            Ob[srow * (NH_ * HD_) + h * HD_ + col] = f2b(o[j][e] * inv);
        }
    }
}

// ---------------------------------------------------------------------------
extern "C" void kernel_launch(void* const* d_in, const int* in_sizes, int n_in,
                              void* d_out, int out_size, void* d_ws, size_t ws_size,
                              hipStream_t stream) {
    (void)in_sizes; (void)n_in; (void)out_size; (void)ws_size;
    const float* x   = (const float*)d_in[0];
    /* d_in[1] = attention_mask: pure causal, applied analytically in-kernel */
    const int*   pos = (const int*)d_in[2];
    const float* Wq  = (const float*)d_in[3];
    const float* Wk  = (const float*)d_in[4];
    const float* Wv  = (const float*)d_in[5];
    const float* Wo  = (const float*)d_in[6];
    float* out = (float*)d_out;

    char* ws = (char*)d_ws;
    size_t off = 0;
    auto take = [&](size_t bytes) { char* p = ws + off; off += bytes; return p; };
    unsigned short* xb   = (unsigned short*)take((size_t)B_ * S_ * H_ * 2);
    unsigned short* wqt  = (unsigned short*)take((size_t)H_ * (NH_  * HD_) * 2);  // [2048][2048]
    unsigned short* wkt  = (unsigned short*)take((size_t)H_ * (NKV_ * HD_) * 2);  // [512][2048]
    unsigned short* wvt  = (unsigned short*)take((size_t)H_ * (NKV_ * HD_) * 2);  // [512][2048]
    unsigned short* wot  = (unsigned short*)take((size_t)H_ * H_ * 2);            // [2048][2048]
    unsigned short* qb   = (unsigned short*)take((size_t)B_ * S_ * NH_  * HD_ * 2);
    unsigned short* kbuf = (unsigned short*)take((size_t)B_ * S_ * NKV_ * HD_ * 2);
    unsigned short* vtg  = (unsigned short*)take((size_t)B_ * S_ * NKV_ * HD_ * 2); // [512][B*S]
    unsigned short* atnb = (unsigned short*)take((size_t)B_ * S_ * H_ * 2);

    const int CT = 256;
    {
        size_t n = (size_t)B_ * S_ * H_;
        cvt_f32_bf16<<<(unsigned)((n + CT - 1) / CT), CT, 0, stream>>>(x, xb, n);
    }
    // One-time weight transpose+convert: W[K][N] f32 -> Wt[N][K] bf16
    cvt_transpose<<<dim3((NH_  * HD_) / 32, H_ / 32), 256, 0, stream>>>(Wq, wqt, H_, NH_  * HD_);
    cvt_transpose<<<dim3((NKV_ * HD_) / 32, H_ / 32), 256, 0, stream>>>(Wk, wkt, H_, NKV_ * HD_);
    cvt_transpose<<<dim3((NKV_ * HD_) / 32, H_ / 32), 256, 0, stream>>>(Wv, wvt, H_, NKV_ * HD_);
    cvt_transpose<<<dim3(H_ / 32,           H_ / 32), 256, 0, stream>>>(Wo, wot, H_, H_);

    const int M = B_ * S_;  // 4096
    gemm_bf16<false, false><<<dim3((NH_  * HD_) / 128, M / 128), 256, 0, stream>>>(
        xb, wqt, qb,   M, NH_  * HD_, H_);
    gemm_bf16<false, false><<<dim3((NKV_ * HD_) / 128, M / 128), 256, 0, stream>>>(
        xb, wkt, kbuf, M, NKV_ * HD_, H_);
    gemm_bf16<false, true><<<dim3((NKV_ * HD_) / 128, M / 128), 256, 0, stream>>>(
        xb, wvt, vtg,  M, NKV_ * HD_, H_);   // V written pre-transposed [512][M]

    {
        size_t nq = (size_t)B_ * S_ * NH_  * (HD_ / 2);
        size_t nk = (size_t)B_ * S_ * NKV_ * (HD_ / 2);
        rope_kernel<<<(unsigned)((nq + CT - 1) / CT), CT, 0, stream>>>(qb,   pos, NH_);
        rope_kernel<<<(unsigned)((nk + CT - 1) / CT), CT, 0, stream>>>(kbuf, pos, NKV_);
    }

    attn_kernel<<<dim3(S_ / 64, NH_, B_), 128, 0, stream>>>(qb, kbuf, vtg, atnb);

    gemm_bf16<true, false><<<dim3(H_ / 128, M / 128), 256, 0, stream>>>(
        atnb, wot, out, M, H_, H_);
}
